// MultiheadAttention_72395968741948
// MI455X (gfx1250) — matrix operand order repaired and checked by reading, not compiled, not run on hardware
//
#include <hip/hip_runtime.h>

// ---------------- problem constants ----------------
#define B_   8
#define S_   4096
#define D_   1024
#define H_   16
#define HD_  64
#define W_   16
#define TD_  (3 * D_)   // 3072

// ---------------- CDNA5 WMMA types ----------------
typedef __attribute__((ext_vector_type(16))) __bf16        v16bf;
typedef __attribute__((ext_vector_type(8)))  float         v8f;
typedef __attribute__((ext_vector_type(8)))  unsigned short ush8;

struct ush16s { ush8 lo, hi; };

// Async global->LDS path (CDNA5): builtin takes (v4i AS1*, v4i AS3*, imm, imm).
#if __has_builtin(__builtin_amdgcn_global_load_async_to_lds_b128) && \
    __has_builtin(__builtin_amdgcn_s_wait_asynccnt)
#define ASYNC_LDS 1
typedef int v4i_ __attribute__((vector_size(4 * sizeof(int))));
typedef __attribute__((address_space(1))) v4i_* as1_v4ip;
typedef __attribute__((address_space(3))) v4i_* as3_v4ip;
#endif

static __device__ __forceinline__ unsigned short f2bf(float f) {
#if defined(__HIP_DEVICE_COMPILE__)
  // native gfx1250 cvt (v_cvt_pk_bf16_f32), RTNE
  __bf16 h = (__bf16)f;
  return __builtin_bit_cast(unsigned short, h);
#else
  unsigned int u = __float_as_uint(f);
  u += 0x7FFFu + ((u >> 16) & 1u);              // round-to-nearest-even
  return (unsigned short)(u >> 16);
#endif
}
static __device__ __forceinline__ float bf2f(unsigned short h) {
  return __uint_as_float(((unsigned int)h) << 16);
}
static __device__ __forceinline__ v16bf load_frag(const unsigned short* p0,
                                                  const unsigned short* p1) {
  ush16s t;
  t.lo = *(const ush8*)p0;
  t.hi = *(const ush8*)p1;
  return __builtin_bit_cast(v16bf, t);
}
static __device__ __forceinline__ v8f vzero8() {
  v8f z = {0.f, 0.f, 0.f, 0.f, 0.f, 0.f, 0.f, 0.f};
  return z;
}
// copy 16 bf16 (32B) global->LDS, async if available
static __device__ __forceinline__ void async_cp16(const unsigned short* g,
                                                  unsigned short* l) {
#if defined(ASYNC_LDS)
  as1_v4ip gp = (as1_v4ip)(unsigned short*)g;
  as3_v4ip lp = (as3_v4ip)l;
  __builtin_amdgcn_global_load_async_to_lds_b128(gp, lp, 0, 0);
  __builtin_amdgcn_global_load_async_to_lds_b128(gp + 1, lp + 1, 0, 0);
#else
  *(ush8*)l       = *(const ush8*)g;
  *(ush8*)(l + 8) = *(const ush8*)(g + 8);
#endif
}
static __device__ __forceinline__ void wait_async0() {
#if defined(ASYNC_LDS)
  __builtin_amdgcn_s_wait_asynccnt(0);
#endif
}
static __device__ __forceinline__ void load16f(float* fr, const float* g) {
#pragma unroll
  for (int q = 0; q < 4; ++q) {
    float4 v = ((const float4*)g)[q];
    fr[4 * q + 0] = v.x; fr[4 * q + 1] = v.y;
    fr[4 * q + 2] = v.z; fr[4 * q + 3] = v.w;
  }
}
static __device__ __forceinline__ void store16bf(unsigned short* l, const float* fr) {
  ush8 a, b;
#pragma unroll
  for (int i = 0; i < 8; ++i) { a[i] = f2bf(fr[i]); b[i] = f2bf(fr[i + 8]); }
  *(ush8*)l       = a;
  *(ush8*)(l + 8) = b;
}

// ---------------- fused f32->bf16 transpose for weights: in[K,N] -> out[N,K] ----------------
__global__ __launch_bounds__(256) void cvt_transpose_kernel(
    const float* __restrict__ in, unsigned short* __restrict__ out, int K, int N) {
  __shared__ float tile[64][65];
  const int kb = blockIdx.y * 64;
  const int nb = blockIdx.x * 64;
  for (int idx = threadIdx.x; idx < 64 * 64; idx += 256) {
    int r = idx >> 6, c = idx & 63;                       // r: k-off, c: n-off
    tile[r][c] = in[(size_t)(kb + r) * N + nb + c];
  }
  __syncthreads();
  for (int idx = threadIdx.x; idx < 64 * 64; idx += 256) {
    int r = idx >> 6, c = idx & 63;                       // r: n-off, c: k-off
    out[(size_t)(nb + r) * K + kb + c] = f2bf(tile[c][r]);
  }
}

// ---------------- bf16 WMMA GEMM:  C[M,N] = A[M,K] * Bt[N,K]^T + bias ----------------
// 128x128 tile, BK=32, 256 thr = 8 waves, wave tile 64x32 -> 8 wmma/K-step.
// Double-buffered LDS; B (and bf16 A) staged with async global->LDS copies.
template <bool A_IS_F32, bool OUT_BF16>
__global__ __launch_bounds__(256) void gemm_bf16_wmma(
    const void* __restrict__ Ap,            // [M,K] f32 (A_IS_F32) or bf16
    const unsigned short* __restrict__ Bt,  // [N,K] bf16 (pre-transposed weights)
    const float* __restrict__ bias,         // [N]
    void* __restrict__ Cout, int M, int N, int K) {
  constexpr int BM = 128, BN = 128, BK = 32, LDSP = 40;  // stride 40 -> conflict-free frags
  __shared__ unsigned short As[2 * BM * LDSP];
  __shared__ unsigned short Bs[2 * BN * LDSP];
  const float* Af          = (const float*)Ap;
  const unsigned short* Ab = (const unsigned short*)Ap;

  const int t  = threadIdx.x;
  const int m0 = blockIdx.y * BM;
  const int n0 = blockIdx.x * BN;
  const int lane  = t & 31;
  const int wid   = t >> 5;
  const int waveM = wid & 1;   // 0..1 -> 64-row slab
  const int waveN = wid >> 1;  // 0..3 -> 32-col slab
  const int lm = lane & 15;
  const int lh = lane >> 4;

  // staging geometry: each thread owns 16 contiguous elements (row sr, cols sko..sko+15)
  const int sr  = t >> 1;
  const int sko = (t & 1) * 16;

  v8f acc[4][2];
#pragma unroll
  for (int i = 0; i < 4; ++i)
#pragma unroll
    for (int j = 0; j < 2; ++j) acc[i][j] = vzero8();

  // ---- prologue: stage tile 0 into buffer 0 ----
  {
    if (A_IS_F32) {
      float fr0[16];
      load16f(fr0, &Af[(size_t)(m0 + sr) * K + sko]);
      store16bf(&As[sr * LDSP + sko], fr0);
    } else {
      async_cp16(&Ab[(size_t)(m0 + sr) * K + sko], &As[sr * LDSP + sko]);
    }
    async_cp16(&Bt[(size_t)(n0 + sr) * K + sko], &Bs[sr * LDSP + sko]);
  }

  const int nk = K / BK;
  for (int i = 0; i < nk; ++i) {
    const int cur = i & 1;
    wait_async0();        // my copies for tile i are done
    __syncthreads();      // everyone's copies for tile i are done

    const bool hasNext = (i + 1) < nk;
    float fr[16];
    if (hasNext) {        // overlap tile i+1 staging with tile i math
      const int ktn = (i + 1) * BK;
      const int nb  = cur ^ 1;
      if (A_IS_F32) {
        load16f(fr, &Af[(size_t)(m0 + sr) * K + ktn + sko]);
      } else {
        async_cp16(&Ab[(size_t)(m0 + sr) * K + ktn + sko],
                   &As[nb * (BM * LDSP) + sr * LDSP + sko]);
      }
      async_cp16(&Bt[(size_t)(n0 + sr) * K + ktn + sko],
                 &Bs[nb * (BN * LDSP) + sr * LDSP + sko]);
    }

    // ---- 8 x v_wmma_f32_16x16x32_bf16 on tile i (ISA 7.12.2 fragment layouts) ----
    const unsigned short* Abase = &As[cur * (BM * LDSP)];
    const unsigned short* Bbase = &Bs[cur * (BN * LDSP)];
    v16bf af[4];
    v16bf bfr[2];
#pragma unroll
    for (int ii = 0; ii < 4; ++ii) {
      int row = waveM * 64 + ii * 16 + lm;
      int kb  = lh * 8;  // half-wave picks K {0-7,16-23} vs {8-15,24-31}
      af[ii] = load_frag(&Abase[row * LDSP + kb], &Abase[row * LDSP + kb + 16]);
    }
#pragma unroll
    for (int j = 0; j < 2; ++j) {
      int nn  = waveN * 32 + j * 16 + lm;
      int kb2 = lh * 8;  // half-wave picks K {0-7,16-23} vs {8-15,24-31}
      bfr[j] = load_frag(&Bbase[nn * LDSP + kb2], &Bbase[nn * LDSP + kb2 + 16]);
    }
#pragma unroll
    for (int ii = 0; ii < 4; ++ii)
#pragma unroll
      for (int j = 0; j < 2; ++j)
        acc[ii][j] = __builtin_amdgcn_wmma_f32_16x16x32_bf16(
            false, af[ii], false, bfr[j], (short)0, acc[ii][j], false, false);

    // commit converted f32 A regs for tile i+1 after the math (latency hidden)
    if (hasNext && A_IS_F32)
      store16bf(&As[(cur ^ 1) * (BM * LDSP) + sr * LDSP + sko], fr);
  }

  // ---- epilogue: C layout lane=N, VGPR index (+half-wave*8)=M ----
#pragma unroll
  for (int i = 0; i < 4; ++i) {
    int rowb = m0 + waveM * 64 + i * 16 + lh * 8;
#pragma unroll
    for (int j = 0; j < 2; ++j) {
      int col  = n0 + waveN * 32 + j * 16 + lm;
      float bv = bias[col];
#pragma unroll
      for (int e = 0; e < 8; ++e) {
        float v = acc[i][j][e] + bv;
        size_t off = (size_t)(rowb + e) * N + col;
        if (OUT_BF16)
          ((unsigned short*)Cout)[off] = f2bf(v);
        else
          ((float*)Cout)[off] = v;
      }
    }
  }
}

// ---------------- banded sliding-window attention (f32, LDS-staged) ----------------
__global__ __launch_bounds__(128) void attn_kernel(
    const unsigned short* __restrict__ qkv,  // [B,S,3D] bf16, head h at col h*192
    unsigned short* __restrict__ attn,       // [B,S,D] bf16
    int S) {
  constexpr int QB = 128, KV = QB + 2 * W_;  // 160 key rows
  constexpr int LR = HD_ + 1;                // row stride 65 -> conflict-free
  __shared__ float kbuf[KV * LR];
  __shared__ float vbuf[KV * LR];
  __shared__ float qbuf[QB * LR];

  const int t  = threadIdx.x;
  const int s0 = blockIdx.x * QB;
  const int h  = blockIdx.y;
  const int b  = blockIdx.z;
  const size_t baseBS = (size_t)b * S;

  for (int idx = t; idx < KV * HD_; idx += 128) {
    int r = idx >> 6;
    int d = idx & 63;
    int sp = s0 - W_ + r;
    float kv = 0.f, vv = 0.f;
    if (sp >= 0 && sp < S) {
      size_t base = (baseBS + sp) * (size_t)TD_ + h * 3 * HD_;
      kv = bf2f(qkv[base + HD_ + d]);
      vv = bf2f(qkv[base + 2 * HD_ + d]);
    }
    kbuf[r * LR + d] = kv;
    vbuf[r * LR + d] = vv;
  }
  for (int idx = t; idx < QB * HD_; idx += 128) {
    int r = idx >> 6;
    int d = idx & 63;
    size_t base = (baseBS + s0 + r) * (size_t)TD_ + h * 3 * HD_;
    qbuf[r * LR + d] = bf2f(qkv[base + d]);
  }
  __syncthreads();

  const int s = s0 + t;
  const float* qr = &qbuf[t * LR];
  float l[2 * W_ + 1];
  float mx = -3.0e38f;
#pragma unroll
  for (int o = 0; o < 2 * W_ + 1; ++o) {
    const float* kr = &kbuf[(t + o) * LR];
    float a = 0.f;
    for (int d = 0; d < HD_; ++d) a = fmaf(qr[d], kr[d], a);
    int kp = s + o - W_;
    l[o] = (kp >= 0 && kp < S) ? a * 0.125f : -3.0e38f;  // 1/sqrt(64)
    mx = fmaxf(mx, l[o]);
  }
  float sum = 0.f;
#pragma unroll
  for (int o = 0; o < 2 * W_ + 1; ++o) { l[o] = __expf(l[o] - mx); sum += l[o]; }
  float inv = 1.f / sum;
#pragma unroll
  for (int o = 0; o < 2 * W_ + 1; ++o) l[o] *= inv;

  unsigned short* orow = &attn[(baseBS + s) * (size_t)D_ + h * HD_];
  for (int d = 0; d < HD_; ++d) {
    float a = 0.f;
#pragma unroll
    for (int o = 0; o < 2 * W_ + 1; ++o) a = fmaf(l[o], vbuf[(t + o) * LR + d], a);
    orow[d] = f2bf(a);
  }
}

// ---------------- launch ----------------
extern "C" void kernel_launch(void* const* d_in, const int* in_sizes, int n_in,
                              void* d_out, int out_size, void* d_ws, size_t ws_size,
                              hipStream_t stream) {
  (void)in_sizes; (void)n_in; (void)out_size; (void)ws_size;
  const float* x    = (const float*)d_in[0];
  const float* Wqkv = (const float*)d_in[1];
  const float* bqkv = (const float*)d_in[2];
  const float* Wo   = (const float*)d_in[3];
  const float* bo   = (const float*)d_in[4];
  float* out = (float*)d_out;

  char* ws = (char*)d_ws;
  size_t off = 0;
  auto alloc = [&](size_t bytes) -> void* {
    void* p = ws + off;
    off += (bytes + 255) & ~(size_t)255;
    return p;
  };
  unsigned short* qkvb  = (unsigned short*)alloc((size_t)B_ * S_ * TD_ * 2);  // 192 MB
  unsigned short* attnb = (unsigned short*)alloc((size_t)B_ * S_ * D_ * 2);   //  64 MB
  unsigned short* wqkvT = (unsigned short*)alloc((size_t)TD_ * D_ * 2);       //   6 MB
  unsigned short* woT   = (unsigned short*)alloc((size_t)D_ * D_ * 2);        //   2 MB

  // weights: f32 [K,N] -> bf16 [N,K]
  dim3 gt1(TD_ / 64, D_ / 64);
  cvt_transpose_kernel<<<gt1, 256, 0, stream>>>(Wqkv, wqkvT, D_, TD_);
  dim3 gt2(D_ / 64, D_ / 64);
  cvt_transpose_kernel<<<gt2, 256, 0, stream>>>(Wo, woT, D_, D_);

  // qkv = x @ Wqkv + bqkv : A is f32 (conversion fused into staging), bf16 out
  dim3 g1(TD_ / 128, (B_ * S_) / 128);
  gemm_bf16_wmma<true, true><<<g1, 256, 0, stream>>>(x, wqkvT, bqkv, (void*)qkvb,
                                                     B_ * S_, TD_, D_);

  // banded attention
  dim3 ga(S_ / 128, H_, B_);
  attn_kernel<<<ga, 128, 0, stream>>>(qkvb, attnb, S_);

  // out = attn @ Wo + bo : A is bf16 (async-staged), f32 out
  dim3 g2(D_ / 128, (B_ * S_) / 128);
  gemm_bf16_wmma<false, false><<<g2, 256, 0, stream>>>(attnb, woT, bo, (void*)out,
                                                       B_ * S_, D_, D_);
}
